// DGIWithGIN_25460566130973
// MI455X (gfx1250) — matrix operand order repaired and checked by reading, not compiled, not run on hardware
//
#include <hip/hip_runtime.h>
#include <hip/hip_bf16.h>

// ---------------------------------------------------------------------------
// DGI + 3-layer GIN encoder for MI455X (gfx1250, wave32, WMMA).
// GEMMs use v_wmma_f32_16x16x32_f16 (f16 in / f32 accumulate).
// Fragments are fetched as ds_load_b128 pairs (A row-major, W transposed).
// ---------------------------------------------------------------------------

#define Hdim 128
constexpr int   Nn  = 100000;
constexpr int   Ee  = 600000;
constexpr float EPS = 1e-5f;

typedef __attribute__((ext_vector_type(16))) _Float16 v16h;
typedef __attribute__((ext_vector_type(8)))  _Float16 v8h;
typedef __attribute__((ext_vector_type(4)))  _Float16 v4h;
typedef __attribute__((ext_vector_type(8)))  float    v8f;

#define LDS_PITCH (Hdim + 8)   // 136 halves = 272 B: 16B-aligned, conflict-free

// ---------------------------------------------------------------- utilities
__global__ __launch_bounds__(256) void k_zero(float* __restrict__ p, int n) {
  for (int i = blockIdx.x * blockDim.x + threadIdx.x; i < n;
       i += gridDim.x * blockDim.x) p[i] = 0.f;
}

// out[n][:] = x[perm[n]][:]   (feature-shuffle corruption)
__global__ __launch_bounds__(256)
void k_gather(const float* __restrict__ x, const long long* __restrict__ perm,
              float* __restrict__ out) {
  const int total = Nn * (Hdim / 4);
  for (int i = blockIdx.x * blockDim.x + threadIdx.x; i < total;
       i += gridDim.x * blockDim.x) {
    const int n = i >> 5, q = i & 31;               // 32 float4 per row
    const long long pn = perm[n];
    const float4 v = ((const float4*)(x + pn * (long long)Hdim))[q];
    ((float4*)(out + (long long)n * Hdim))[q] = v;
  }
}

// agg[dst] += h[src]   (edge-parallel, float4 per lane-slot, fp32 atomics)
__global__ __launch_bounds__(256)
void k_agg(const float* __restrict__ h, const long long* __restrict__ ei,
           float* __restrict__ agg) {
  const long long total = (long long)Ee * 32;
  for (long long i = blockIdx.x * (long long)blockDim.x + threadIdx.x; i < total;
       i += (long long)gridDim.x * blockDim.x) {
    const int e = (int)(i >> 5), q = (int)(i & 31);
    const long long s = ei[e], d = ei[Ee + e];
    const float4 v = ((const float4*)(h + s * Hdim))[q];
    float* ap = agg + d * Hdim + q * 4;
    atomicAdd(ap + 0, v.x); atomicAdd(ap + 1, v.y);
    atomicAdd(ap + 2, v.z); atomicAdd(ap + 3, v.w);
  }
}

// ------------------------------------------------------------------- GEMM
// y[N,H] = pre(in) @ W + bias, pre(in) = in0 + in1 | relu(a*in0 + c).
// Fused per-column sum / sum^2 accumulation for the following BatchNorm.
// Each block: 128-row tile; each wave: 16 rows x 128 cols (8 WMMA acc tiles).
__global__ __launch_bounds__(256)
void k_gemm(const float* __restrict__ in0, const float* __restrict__ in1,
            const float* __restrict__ aff, const float* __restrict__ W,
            const float* __restrict__ bias, float* __restrict__ out,
            float* __restrict__ stats) {
  __shared__ __align__(16) _Float16 Xs[128 * LDS_PITCH];       // row-major tile
  __shared__ __align__(16) _Float16 Wt[Hdim * LDS_PITCH];      // TRANSPOSED: Wt[col][k]
  __shared__ float aA[Hdim], aC[Hdim];

  const int tid = threadIdx.x;
  // stage weights (f32 -> f16), transposed so K is contiguous per column
  for (int i = tid; i < Hdim * Hdim / 4; i += 256) {
    const int r = i >> 5, q = i & 31;                 // W row r, cols 4q..4q+3
    const float4 w = ((const float4*)(W + r * Hdim))[q];
    Wt[(q * 4 + 0) * LDS_PITCH + r] = (_Float16)w.x;
    Wt[(q * 4 + 1) * LDS_PITCH + r] = (_Float16)w.y;
    Wt[(q * 4 + 2) * LDS_PITCH + r] = (_Float16)w.z;
    Wt[(q * 4 + 3) * LDS_PITCH + r] = (_Float16)w.w;
  }
  if (aff) for (int i = tid; i < Hdim; i += 256) { aA[i] = aff[i]; aC[i] = aff[Hdim + i]; }

  const int lane   = tid & 31;
  const int wave   = tid >> 5;      // 0..7
  const int laneHi = lane >> 4;     // 0/1
  const int lcol   = lane & 15;
  const int nRowBlocks = (Nn + 127) / 128;

  for (int rb = blockIdx.x; rb < nRowBlocks; rb += gridDim.x) {
    const long long row0 = (long long)rb * 128;
    __syncthreads();                       // previous tile fully consumed
    // stage activation tile (with fused pre-op) as f16, ds_store_b64 packed
    for (int i = tid; i < 128 * 32; i += 256) {
      const int r = i >> 5, q = i & 31;
      const long long row = row0 + r;
      float4 v = make_float4(0.f, 0.f, 0.f, 0.f);
      if (row < Nn) {
        v = ((const float4*)(in0 + row * Hdim))[q];
        if (in1) {
          const float4 u = ((const float4*)(in1 + row * Hdim))[q];
          v.x += u.x; v.y += u.y; v.z += u.z; v.w += u.w;
        }
        if (aff) {
          const int c = q * 4;
          v.x = fmaxf(aA[c + 0] * v.x + aC[c + 0], 0.f);
          v.y = fmaxf(aA[c + 1] * v.y + aC[c + 1], 0.f);
          v.z = fmaxf(aA[c + 2] * v.z + aC[c + 2], 0.f);
          v.w = fmaxf(aA[c + 3] * v.w + aC[c + 3], 0.f);
        }
      }
      const v4h hv = { (_Float16)v.x, (_Float16)v.y, (_Float16)v.z, (_Float16)v.w };
      *(v4h*)&Xs[r * LDS_PITCH + q * 4] = hv;
    }
    __builtin_prefetch(in0 + (row0 + 128 * (long long)gridDim.x) * Hdim, 0, 1);
    __syncthreads();

    const int rowBase = wave * 16;
    v8f acc[8];
#pragma unroll
    for (int t = 0; t < 8; ++t) acc[t] = (v8f){0.f,0.f,0.f,0.f,0.f,0.f,0.f,0.f};

    const v8h* Ap = (const v8h*)&Xs[(rowBase + lcol) * LDS_PITCH];

#pragma unroll
    for (int ks = 0; ks < 4; ++ks) {
      // A fragment: a[0..7]=K kb+8*laneHi+(0..7); a[8..15]=K kb+16+8*laneHi+(0..7)
      const v8h alo = Ap[4 * ks + laneHi];
      const v8h ahi = Ap[4 * ks + 2 + laneHi];
      const v16h aF = __builtin_shufflevector(alo, ahi,
          0, 1, 2, 3, 4, 5, 6, 7, 8, 9, 10, 11, 12, 13, 14, 15);
#pragma unroll
      for (int t = 0; t < 8; ++t) {
        const int col = t * 16 + lcol;
        // B fragment: b[0..7]=K kb+8*laneHi+(0..7); b[8..15]=K kb+16+8*laneHi+(0..7) along column `col`
        const v8h* Bp = (const v8h*)&Wt[col * LDS_PITCH];
        const v8h blo = Bp[4 * ks + laneHi];
        const v8h bhi = Bp[4 * ks + 2 + laneHi];
        const v16h bF = __builtin_shufflevector(blo, bhi,
            0, 1, 2, 3, 4, 5, 6, 7, 8, 9, 10, 11, 12, 13, 14, 15);
        acc[t] = __builtin_amdgcn_wmma_f32_16x16x32_f16(
            false, aF, false, bF, (short)0, acc[t], false, false);
      }
    }

    // epilogue: + bias, store y, fused column stats (sum, sum^2)
#pragma unroll
    for (int t = 0; t < 8; ++t) {
      const int col = t * 16 + lcol;
      const float bc = bias[col];
      float s = 0.f, ss = 0.f;
#pragma unroll
      for (int v = 0; v < 8; ++v) {
        const long long row = row0 + rowBase + v + 8 * laneHi;
        const float val = acc[t][v] + bc;
        if (row < Nn) {
          out[row * Hdim + col] = val;
          s += val; ss += val * val;
        }
      }
      s  += __shfl_xor(s, 16, 32);       // merge the two laneHi halves
      ss += __shfl_xor(ss, 16, 32);
      if (laneHi == 0 && stats) {
        atomicAdd(&stats[col], s);
        atomicAdd(&stats[Hdim + col], ss);
      }
    }
  }
}

// stats -> affine:  a = g * rsqrt(var+eps), c = be - mu*a
__global__ __launch_bounds__(128)
void k_bn_finalize(const float* __restrict__ stats, const float* __restrict__ g,
                   const float* __restrict__ be, float* __restrict__ aff, float invN) {
  const int c = threadIdx.x;
  const float mu  = stats[c] * invN;
  const float var = stats[Hdim + c] * invN - mu * mu;
  const float a   = g[c] * rsqrtf(var + EPS);
  aff[c] = a;
  aff[Hdim + c] = be[c] - mu * a;
}

// z = maybe_relu(a*y + c); optional fused column stats of z
__global__ __launch_bounds__(128)
void k_bn_apply(const float* __restrict__ y, const float* __restrict__ aff,
                int doRelu, float* __restrict__ z, float* __restrict__ stats) {
  const int c = threadIdx.x;                 // one column per thread (coalesced)
  const float a = aff[c], b = aff[Hdim + c];
  float s = 0.f, ss = 0.f;
  for (long long row = blockIdx.x; row < Nn; row += gridDim.x) {
    float v = a * y[row * Hdim + c] + b;
    if (doRelu) v = fmaxf(v, 0.f);
    z[row * Hdim + c] = v;
    s += v; ss += v * v;
  }
  if (stats) { atomicAdd(&stats[c], s); atomicAdd(&stats[Hdim + c], ss); }
}

__global__ __launch_bounds__(128)
void k_colsum(const float* __restrict__ h, float* __restrict__ stats) {
  const int c = threadIdx.x;
  float s = 0.f;
  for (long long row = blockIdx.x; row < Nn; row += gridDim.x)
    s += h[row * Hdim + c];
  atomicAdd(&stats[c], s);
}

// summary MLP + bilinear vector: v = Wbil @ (relu(mean@Ws1+bs1)@Ws2+bs2)
__global__ __launch_bounds__(128)
void k_head(const float* __restrict__ colsum,
            const float* __restrict__ Ws1, const float* __restrict__ bs1,
            const float* __restrict__ Ws2, const float* __restrict__ bs2,
            const float* __restrict__ Wbil, float* __restrict__ vout) {
  __shared__ float sm[Hdim], t1[Hdim], s2[Hdim];
  const int j = threadIdx.x;
  sm[j] = colsum[j] * (1.f / (float)Nn);
  __syncthreads();
  float acc = bs1[j];
  for (int i = 0; i < Hdim; ++i) acc += sm[i] * Ws1[i * Hdim + j];
  t1[j] = fmaxf(acc, 0.f);
  __syncthreads();
  acc = bs2[j];
  for (int i = 0; i < Hdim; ++i) acc += t1[i] * Ws2[i * Hdim + j];
  s2[j] = acc;
  __syncthreads();
  acc = 0.f;
  for (int i = 0; i < Hdim; ++i) acc += Wbil[j * Hdim + i] * s2[i];
  vout[j] = acc;
}

// scores[row] = h[row] . v + bbil      (one wave per row, shuffle reduction)
__global__ __launch_bounds__(256)
void k_scores(const float* __restrict__ h, const float* __restrict__ v,
              const float* __restrict__ bbil, float* __restrict__ out) {
  const int lane = threadIdx.x & 31;
  const int wid  = blockIdx.x * (blockDim.x >> 5) + (threadIdx.x >> 5);
  const int nw   = gridDim.x * (blockDim.x >> 5);
  const float4 vv = ((const float4*)v)[lane];
  const float bb  = bbil[0];
  for (long long row = wid; row < Nn; row += nw) {
    const float4 xv = ((const float4*)(h + row * Hdim))[lane];
    float p = xv.x * vv.x + xv.y * vv.y + xv.z * vv.z + xv.w * vv.w;
    p += __shfl_xor(p, 16, 32);
    p += __shfl_xor(p,  8, 32);
    p += __shfl_xor(p,  4, 32);
    p += __shfl_xor(p,  2, 32);
    p += __shfl_xor(p,  1, 32);
    if (lane == 0) out[row] = p + bb;
  }
}

// ---------------------------------------------------------------------------
extern "C" void kernel_launch(void* const* d_in, const int* in_sizes, int n_in,
                              void* d_out, int out_size, void* d_ws, size_t ws_size,
                              hipStream_t stream) {
  const float*      x    = (const float*)d_in[0];
  const long long*  ei   = (const long long*)d_in[1];
  const long long*  perm = (const long long*)d_in[2];

  const float *W1[3], *b1[3], *g1[3], *be1[3], *W2[3], *b2[3], *g2[3], *be2[3],
              *gn[3], *bn[3];
  const float *Ws1, *bs1, *Ws2, *bs2, *Wbil, *bbil;

  // param flattening order: JAX pytree (sorted keys) vs dict-insertion order.
  const bool sorted = (n_in > 4 && in_sizes[4] == Hdim * Hdim);
  if (sorted) {  // Wbil,Ws1,Ws2,bbil,bs1,bs2, conv{W1,W2,b1,b2,be1,be2,bn,g1,g2,gn}
    Wbil = (const float*)d_in[3]; Ws1 = (const float*)d_in[4];
    Ws2  = (const float*)d_in[5]; bbil = (const float*)d_in[6];
    bs1  = (const float*)d_in[7]; bs2 = (const float*)d_in[8];
    for (int l = 0; l < 3; ++l) {
      const int b = 9 + 10 * l;
      W1[l]  = (const float*)d_in[b + 0]; W2[l]  = (const float*)d_in[b + 1];
      b1[l]  = (const float*)d_in[b + 2]; b2[l]  = (const float*)d_in[b + 3];
      be1[l] = (const float*)d_in[b + 4]; be2[l] = (const float*)d_in[b + 5];
      bn[l]  = (const float*)d_in[b + 6]; g1[l]  = (const float*)d_in[b + 7];
      g2[l]  = (const float*)d_in[b + 8]; gn[l]  = (const float*)d_in[b + 9];
    }
  } else {       // insertion order
    for (int l = 0; l < 3; ++l) {
      const int b = 3 + 10 * l;
      W1[l]  = (const float*)d_in[b + 0]; b1[l]  = (const float*)d_in[b + 1];
      g1[l]  = (const float*)d_in[b + 2]; be1[l] = (const float*)d_in[b + 3];
      W2[l]  = (const float*)d_in[b + 4]; b2[l]  = (const float*)d_in[b + 5];
      g2[l]  = (const float*)d_in[b + 6]; be2[l] = (const float*)d_in[b + 7];
      gn[l]  = (const float*)d_in[b + 8]; bn[l]  = (const float*)d_in[b + 9];
    }
    Ws1 = (const float*)d_in[33]; bs1 = (const float*)d_in[34];
    Ws2 = (const float*)d_in[35]; bs2 = (const float*)d_in[36];
    Wbil = (const float*)d_in[37]; bbil = (const float*)d_in[38];
  }

  // workspace: 3 big [N,H] buffers + small stats/affine/vector region
  float* ws   = (float*)d_ws;
  const size_t big = (size_t)Nn * Hdim;
  float* buf0 = ws;                 // h (layer features)
  float* buf1 = ws + big;           // agg / y2
  float* buf2 = ws + 2 * big;       // y1 / z
  float* stA  = ws + 3 * big;       // [2H] sum / sumsq
  float* affA = stA + 2 * Hdim;     // [2H] a / c
  float* vvec = affA + 2 * Hdim;    // [H]
  float* outF = (float*)d_out;

  const float invN = 1.f / (float)Nn;
  const int gemmGrid = (Nn + 127) / 128;

  auto encode = [&](const float* h0) {
    const float* h = h0;
    for (int l = 0; l < 3; ++l) {
      k_zero<<<2048, 256, 0, stream>>>(buf1, Nn * Hdim);
      k_zero<<<1, 256, 0, stream>>>(stA, 2 * Hdim);
      k_agg<<<4096, 256, 0, stream>>>(h, ei, buf1);
      // y1 = (h + agg) @ W1 + b1     (stats -> BN1)
      k_gemm<<<gemmGrid, 256, 0, stream>>>(h, buf1, nullptr, W1[l], b1[l], buf2, stA);
      k_bn_finalize<<<1, 128, 0, stream>>>(stA, g1[l], be1[l], affA, invN);
      k_zero<<<1, 256, 0, stream>>>(stA, 2 * Hdim);
      // y2 = relu(BN1(y1)) @ W2 + b2 (stats -> BN2)
      k_gemm<<<gemmGrid, 256, 0, stream>>>(buf2, nullptr, affA, W2[l], b2[l], buf1, stA);
      k_bn_finalize<<<1, 128, 0, stream>>>(stA, g2[l], be2[l], affA, invN);
      k_zero<<<1, 256, 0, stream>>>(stA, 2 * Hdim);
      // z = maybe_relu(BN2(y2)) with stats for outer BN
      k_bn_apply<<<1024, 128, 0, stream>>>(buf1, affA, l < 2 ? 1 : 0, buf2, stA);
      k_bn_finalize<<<1, 128, 0, stream>>>(stA, gn[l], bn[l], affA, invN);
      // h_next = maybe_relu(outerBN(z))
      k_bn_apply<<<1024, 128, 0, stream>>>(buf2, affA, l < 2 ? 1 : 0, buf0, nullptr);
      h = buf0;
    }
  };

  // ---- positive branch (summary + bilinear vector come from it)
  encode(x);
  k_zero<<<1, 256, 0, stream>>>(stA, 2 * Hdim);
  k_colsum<<<1024, 128, 0, stream>>>(buf0, stA);
  k_head<<<1, 128, 0, stream>>>(stA, Ws1, bs1, Ws2, bs2, Wbil, vvec);
  k_scores<<<1024, 256, 0, stream>>>(buf0, vvec, bbil, outF);

  // ---- negative branch (feature shuffle), reuses the same buffers
  k_gather<<<2048, 256, 0, stream>>>(x, perm, buf0);
  encode(buf0);
  k_scores<<<1024, 256, 0, stream>>>(buf0, vvec, bbil, outF + Nn);

  (void)ws_size; (void)out_size; (void)in_sizes;
}
